// AttentionLayer_42030549959205
// MI455X (gfx1250) — matrix operand order repaired and checked by reading, not compiled, not run on hardware
//
#include <hip/hip_runtime.h>
#include <hip/hip_bf16.h>
#include <cmath>

typedef _Float16 f16;
typedef __attribute__((ext_vector_type(16))) _Float16 v16h;
typedef __attribute__((ext_vector_type(8)))  _Float16 v8h;
typedef __attribute__((ext_vector_type(8)))  float    v8f;

#define B_  32
#define S_  2048
#define D_  1024
#define TY  32                 // y rows per tile
#define XT  64                 // x columns per outer iteration (4 waves x 16)
#define KC  256                // K chunk staged in LDS for X
#define NXT (S_/XT)            // 32
#define NKC (D_/KC)            // 4
#define NCH (NXT*NKC)          // 128 chunks per pass
#define YT_PER_WG 4
#define NGY (S_/TY/YT_PER_WG)  // 16 workgroups per batch in y
#define SCALE_INV (1.0f/32.0f) // 1/sqrt(D)

// ---------------------------------------------------------------------------
// ybar[b,d] = sum_y Y[b,y,d]
__global__ void ybar_kernel(const float* __restrict__ Y, float* __restrict__ ybar) {
    int i = blockIdx.x * blockDim.x + threadIdx.x;   // B_*D_ threads
    int b = i / D_, d = i % D_;
    const float* p = Y + (size_t)b * S_ * D_ + d;
    float s = 0.f;
    for (int y = 0; y < S_; ++y) s += p[(size_t)y * D_];
    ybar[i] = s;
}

// visual[b,x] = ybar[b,:] . X[b,x,:] / (Sy*sqrt(D));  one wave per x row
__global__ void visual_kernel(const float* __restrict__ X,
                              const float* __restrict__ ybar,
                              float* __restrict__ vis) {
    int wid = threadIdx.x >> 5, lane = threadIdx.x & 31;
    long row = (long)blockIdx.x * 8 + wid;           // 0 .. B_*S_-1
    int b = (int)(row / S_);
    int x = (int)(row % S_);
    const float* xp = X + ((size_t)b * S_ + x) * D_;
    const float* yp = ybar + (size_t)b * D_;
    float s = 0.f;
    for (int k = lane; k < D_; k += 32) s += xp[k] * yp[k];
    for (int off = 16; off; off >>= 1) s += __shfl_xor(s, off, 32);
    if (lane == 0) vis[row] = s * (SCALE_INV / (float)S_);
}

// X (f32) -> X16 (f16) streaming conversion, 8 elements per thread
__global__ void cvt_kernel(const float* __restrict__ X, f16* __restrict__ X16) {
    size_t i = (size_t)blockIdx.x * 256 + threadIdx.x;   // N/8 threads
    const float4* xp = (const float4*)X;
    float4 a = xp[i * 2], b = xp[i * 2 + 1];
    v8h h = {(f16)a.x, (f16)a.y, (f16)a.z, (f16)a.w,
             (f16)b.x, (f16)b.y, (f16)b.z, (f16)b.w};
    *(v8h*)(X16 + i * 8) = h;
}

// ---------------------------------------------------------------------------
// Shared WMMA micro-step: one 16x16 score tile K-chunk (8 x 16x16x32 WMMA)
__device__ __forceinline__ v8f wmma_chunk(const f16 yS[TY][D_], const f16* __restrict__ xbuf,
                                          int rowA, int rowB, int hi, int kc, v8f c) {
#pragma unroll
    for (int kk = 0; kk < KC; kk += 32) {
        int kg = kc * KC + kk;
        // A (16x32 f16): lane<16 holds K {0..7,16..23}, lane>=16 {8..15,24..31}
        v8h alo = *(const v8h*)&yS[rowA][kg + hi * 8];
        v8h ahi = *(const v8h*)&yS[rowA][kg + 16 + hi * 8];
        // B (32x16 f16): lane<16 holds K {0..7,16..23}, lane>=16 {8..15,24..31}
        v8h blo = *(const v8h*)&xbuf[rowB * KC + kk + hi * 8];
        v8h bhi = *(const v8h*)&xbuf[rowB * KC + kk + 16 + hi * 8];
        v16h a  = __builtin_shufflevector(alo, ahi, 0,1,2,3,4,5,6,7,8,9,10,11,12,13,14,15);
        v16h bb = __builtin_shufflevector(blo, bhi, 0,1,2,3,4,5,6,7,8,9,10,11,12,13,14,15);
        c = __builtin_amdgcn_wmma_f32_16x16x32_f16(false, a, false, bb, (short)0, c, false, false);
    }
    return c;
}

// Async DMA: one (xt,kc) chunk of f16 X (64 rows x 256 K = 32KB) global -> LDS.
// 2048 x 16B segments, 8 per thread.  Tracked by ASYNCcnt.
__device__ __forceinline__ void issue_chunk(const f16* __restrict__ Xb16, f16* ldsbuf,
                                            int xt, int kc, int tid) {
    unsigned ldsbase = (unsigned)(size_t)ldsbuf;
    const char* gb = (const char*)Xb16;
#pragma unroll
    for (int i = 0; i < 8; ++i) {
        int seg = tid + 256 * i;                 // 0..2047
        int r = seg >> 5, s8 = seg & 31;
        unsigned goff  = (unsigned)((((xt * XT + r) * D_ + kc * KC) << 1) + s8 * 16);
        unsigned laddr = ldsbase + seg * 16;
        asm volatile("global_load_async_to_lds_b128 %0, %1, %2 offset:0"
                     :: "v"(laddr), "v"(goff), "s"(gb) : "memory");
    }
}

__device__ __forceinline__ void wait_async0() {
    asm volatile("s_wait_asynccnt 0" ::: "memory");
}

// ---------------------------------------------------------------------------
// Main WMMA kernel, async-DMA variant: X already converted to f16 in ws.
// For each (b, group-of-4 y-tiles): partial[b,g,x] = sum over its 128 y rows
// of exp(S[y,x]-m_y)/l_y.  Two passes per 32-row y-tile.
__global__ __launch_bounds__(256)
void attn_main_async_kernel(const f16* __restrict__ X16, const float* __restrict__ Y,
                            float* __restrict__ partial) {
    __shared__ f16   yS[TY][D_];           // 64 KB
    __shared__ f16   xA[2][XT * KC];       // 2 x 32 KB ping-pong, DMA-filled
    __shared__ float wstats[2][4][16][2];
    __shared__ float stats[TY][2];
    __shared__ float cacc[2][XT];
    __shared__ float colsum[S_];           // 8 KB

    const int tid  = threadIdx.x;
    const int lane = tid & 31, wid = tid >> 5;
    const int q    = wid >> 1;
    const int ysub = wid & 1;
    const int n    = lane & 15;
    const int hi   = lane >> 4;
    const int b    = blockIdx.x / NGY;
    const int g    = blockIdx.x % NGY;
    const f16*   Xb16 = X16 + (size_t)b * S_ * D_;
    const float* Yb   = Y   + (size_t)b * S_ * D_;

    for (int x = tid; x < S_; x += 256) colsum[x] = 0.f;

    const int rowA = ysub * 16 + n;
    const int rowB = q * 16 + n;

    for (int yt = 0; yt < YT_PER_WG; ++yt) {
        const int y0 = (g * YT_PER_WG + yt) * TY;
        __syncthreads();
        for (int idx = tid; idx < TY * D_; idx += 256) {
            int r = idx >> 10, k = idx & (D_ - 1);
            yS[r][k] = (f16)Yb[(size_t)(y0 + r) * D_ + k];
        }

        // ---------------- pass 1: per-row max / sum-exp ----------------
        float mrun[8], srun[8];
        for (int v = 0; v < 8; ++v) { mrun[v] = -1e30f; srun[v] = 0.f; }
        v8f c = {0.f,0.f,0.f,0.f,0.f,0.f,0.f,0.f};
        issue_chunk(Xb16, xA[0], 0, 0, tid);                 // prologue DMA
        for (int ch = 0; ch < NCH; ++ch) {
            const int xt = ch >> 2, kc = ch & 3, buf = ch & 1;
            wait_async0();                                    // own DMAs into xA[buf] done
            __syncthreads();                                  // everyone's DMAs + old reads done
            if (ch + 1 < NCH)
                issue_chunk(Xb16, xA[buf ^ 1], (ch + 1) >> 2, (ch + 1) & 3, tid);
            c = wmma_chunk(yS, xA[buf], rowA, rowB, hi, kc, c);
            if (kc == NKC - 1) {
#pragma unroll
                for (int v = 0; v < 8; ++v) {
                    float sv = c[v] * SCALE_INV;
                    float mn = fmaxf(mrun[v], sv);
                    srun[v] = srun[v] * __expf(mrun[v] - mn) + __expf(sv - mn);
                    mrun[v] = mn;
                }
                c = (v8f){0.f,0.f,0.f,0.f,0.f,0.f,0.f,0.f};
            }
        }
#pragma unroll
        for (int v = 0; v < 8; ++v) {
            float m = mrun[v], s = srun[v];
            for (int off = 1; off < 16; off <<= 1) {
                float mo = __shfl_xor(m, off, 32);
                float so = __shfl_xor(s, off, 32);
                float mn = fmaxf(m, mo);
                s = s * __expf(m - mn) + so * __expf(mo - mn);
                m = mn;
            }
            if (n == 0) {
                wstats[ysub][q][v + hi * 8][0] = m;
                wstats[ysub][q][v + hi * 8][1] = s;
            }
        }
        __syncthreads();
        if (tid < TY) {
            int ys = tid >> 4, r = tid & 15;
            float mf = -1e30f;
            for (int qq = 0; qq < 4; ++qq) mf = fmaxf(mf, wstats[ys][qq][r][0]);
            float lf = 0.f;
            for (int qq = 0; qq < 4; ++qq)
                lf += wstats[ys][qq][r][1] * __expf(wstats[ys][qq][r][0] - mf);
            stats[tid][0] = mf;
            stats[tid][1] = 1.0f / lf;
        }
        __syncthreads();

        // ---------------- pass 2: column sums of softmax ----------------
        c = (v8f){0.f,0.f,0.f,0.f,0.f,0.f,0.f,0.f};
        issue_chunk(Xb16, xA[0], 0, 0, tid);
        for (int ch = 0; ch < NCH; ++ch) {
            const int xt = ch >> 2, kc = ch & 3, buf = ch & 1;
            wait_async0();
            __syncthreads();
            if (ch + 1 < NCH)
                issue_chunk(Xb16, xA[buf ^ 1], (ch + 1) >> 2, (ch + 1) & 3, tid);
            c = wmma_chunk(yS, xA[buf], rowA, rowB, hi, kc, c);
            if (kc == NKC - 1) {
                float cs = 0.f;
#pragma unroll
                for (int v = 0; v < 8; ++v) {
                    int row = ysub * 16 + v + hi * 8;
                    cs += __expf(c[v] * SCALE_INV - stats[row][0]) * stats[row][1];
                }
                cs += __shfl_xor(cs, 16, 32);
                if (hi == 0) cacc[ysub][q * 16 + n] = cs;
                __syncthreads();
                if (tid < XT) colsum[xt * XT + tid] += cacc[0][tid] + cacc[1][tid];
                c = (v8f){0.f,0.f,0.f,0.f,0.f,0.f,0.f,0.f};
            }
        }
    }
    __syncthreads();
    float* outp = partial + ((size_t)b * NGY + g) * S_;
    for (int x = tid; x < S_; x += 256) outp[x] = colsum[x];
}

// ---------------------------------------------------------------------------
// Fallback variant (no f16 X copy in ws): register-staged X with f32->f16 cvt.
__global__ __launch_bounds__(256)
void attn_main_kernel(const float* __restrict__ X, const float* __restrict__ Y,
                      float* __restrict__ partial) {
    __shared__ f16   yS[TY][D_];
    __shared__ f16   xS[XT * KC];
    __shared__ float wstats[2][4][16][2];
    __shared__ float stats[TY][2];
    __shared__ float cacc[2][XT];
    __shared__ float colsum[S_];

    const int tid  = threadIdx.x;
    const int lane = tid & 31, wid = tid >> 5;
    const int q    = wid >> 1;
    const int ysub = wid & 1;
    const int n    = lane & 15;
    const int hi   = lane >> 4;
    const int b    = blockIdx.x / NGY;
    const int g    = blockIdx.x % NGY;
    const float* Xb = X + (size_t)b * S_ * D_;
    const float* Yb = Y + (size_t)b * S_ * D_;

    for (int x = tid; x < S_; x += 256) colsum[x] = 0.f;

    const int rowA = ysub * 16 + n;
    const int rowB = q * 16 + n;

    for (int yt = 0; yt < YT_PER_WG; ++yt) {
        const int y0 = (g * YT_PER_WG + yt) * TY;
        __syncthreads();
        for (int idx = tid; idx < TY * D_; idx += 256) {
            int r = idx >> 10, k = idx & (D_ - 1);
            yS[r][k] = (f16)Yb[(size_t)(y0 + r) * D_ + k];
        }

        float mrun[8], srun[8];
        for (int v = 0; v < 8; ++v) { mrun[v] = -1e30f; srun[v] = 0.f; }
        for (int xt = 0; xt < NXT; ++xt) {
            v8f c = {0.f,0.f,0.f,0.f,0.f,0.f,0.f,0.f};
            for (int kc = 0; kc < NKC; ++kc) {
                __syncthreads();
                for (int idx = tid; idx < XT * KC; idx += 256) {
                    int r = idx >> 8, k = idx & (KC - 1);
                    xS[r * KC + k] = (f16)Xb[(size_t)(xt * XT + r) * D_ + kc * KC + k];
                }
                __syncthreads();
                c = wmma_chunk(yS, xS, rowA, rowB, hi, kc, c);
            }
            for (int v = 0; v < 8; ++v) {
                float sv = c[v] * SCALE_INV;
                float mn = fmaxf(mrun[v], sv);
                srun[v] = srun[v] * __expf(mrun[v] - mn) + __expf(sv - mn);
                mrun[v] = mn;
            }
        }
        for (int v = 0; v < 8; ++v) {
            float m = mrun[v], s = srun[v];
            for (int off = 1; off < 16; off <<= 1) {
                float mo = __shfl_xor(m, off, 32);
                float so = __shfl_xor(s, off, 32);
                float mn = fmaxf(m, mo);
                s = s * __expf(m - mn) + so * __expf(mo - mn);
                m = mn;
            }
            if (n == 0) {
                wstats[ysub][q][v + hi * 8][0] = m;
                wstats[ysub][q][v + hi * 8][1] = s;
            }
        }
        __syncthreads();
        if (tid < TY) {
            int ys = tid >> 4, r = tid & 15;
            float mf = -1e30f;
            for (int qq = 0; qq < 4; ++qq) mf = fmaxf(mf, wstats[ys][qq][r][0]);
            float lf = 0.f;
            for (int qq = 0; qq < 4; ++qq)
                lf += wstats[ys][qq][r][1] * __expf(wstats[ys][qq][r][0] - mf);
            stats[tid][0] = mf;
            stats[tid][1] = 1.0f / lf;
        }
        __syncthreads();

        for (int xt = 0; xt < NXT; ++xt) {
            v8f c = {0.f,0.f,0.f,0.f,0.f,0.f,0.f,0.f};
            for (int kc = 0; kc < NKC; ++kc) {
                __syncthreads();
                for (int idx = tid; idx < XT * KC; idx += 256) {
                    int r = idx >> 8, k = idx & (KC - 1);
                    xS[r * KC + k] = (f16)Xb[(size_t)(xt * XT + r) * D_ + kc * KC + k];
                }
                __syncthreads();
                c = wmma_chunk(yS, xS, rowA, rowB, hi, kc, c);
            }
            float cs = 0.f;
            for (int v = 0; v < 8; ++v) {
                int row = ysub * 16 + v + hi * 8;
                cs += __expf(c[v] * SCALE_INV - stats[row][0]) * stats[row][1];
            }
            cs += __shfl_xor(cs, 16, 32);
            if (hi == 0) cacc[ysub][q * 16 + n] = cs;
            __syncthreads();
            if (tid < XT) colsum[xt * XT + tid] += cacc[0][tid] + cacc[1][tid];
        }
    }
    __syncthreads();
    float* outp = partial + ((size_t)b * NGY + g) * S_;
    for (int x = tid; x < S_; x += 256) outp[x] = colsum[x];
}

// colw[b,x] = sum_g partial[b,g,x]   (fixed-order -> deterministic)
__global__ void colw_kernel(const float* __restrict__ partial, float* __restrict__ colw) {
    int i = blockIdx.x * 256 + threadIdx.x;
    int b = i / S_, x = i % S_;
    float s = 0.f;
    for (int gg = 0; gg < NGY; ++gg) s += partial[((size_t)b * NGY + gg) * S_ + x];
    colw[i] = s;
}

// out[b,d] = (1/Sy) * sum_x colw[b,x] * X[b,x,d]
__global__ void out_kernel(const float* __restrict__ X, const float* __restrict__ colw,
                           float* __restrict__ out) {
    int i = blockIdx.x * 256 + threadIdx.x;
    int b = i / D_, d = i % D_;
    const float* xp = X + (size_t)b * S_ * D_ + d;
    const float* wp = colw + (size_t)b * S_;
    float s = 0.f;
    for (int x = 0; x < S_; ++x) s += wp[x] * xp[(size_t)x * D_];
    out[i] = s * (1.0f / (float)S_);
}

// ---------------------------------------------------------------------------
extern "C" void kernel_launch(void* const* d_in, const int* in_sizes, int n_in,
                              void* d_out, int out_size, void* d_ws, size_t ws_size,
                              hipStream_t stream) {
    (void)in_sizes; (void)n_in; (void)out_size;
    const float* X = (const float*)d_in[0];   // input_x [B,Sx,D]
    const float* Y = (const float*)d_in[1];   // input_y [B,Sy,D]
    float* out = (float*)d_out;               // [visual B*Sx | out B*D]
    float* ws  = (float*)d_ws;
    float* ybar    = ws;                                   // B*D
    float* partial = ybar + (size_t)B_ * D_;               // B*NGY*S
    float* colw    = partial + (size_t)B_ * NGY * S_;      // B*S

    const size_t float_region = ((size_t)B_ * D_ + (size_t)B_ * NGY * S_ + (size_t)B_ * S_) * 4;
    const size_t x16_bytes    = (size_t)B_ * S_ * D_ * 2;  // f16 copy of X
    f16* X16 = (f16*)((char*)d_ws + float_region);

    ybar_kernel  <<<(B_ * D_) / 256, 256, 0, stream>>>(Y, ybar);
    visual_kernel<<<(B_ * S_) / 8,   256, 0, stream>>>(X, ybar, out);

    if (ws_size >= float_region + x16_bytes) {
        // async-DMA path: pre-convert X to f16, stream via GLOBAL_LOAD_ASYNC_TO_LDS
        cvt_kernel<<<(int)(((size_t)B_ * S_ * D_ / 8) / 256), 256, 0, stream>>>(X, X16);
        attn_main_async_kernel<<<B_ * NGY, 256, 0, stream>>>(X16, Y, partial);
    } else {
        attn_main_kernel<<<B_ * NGY, 256, 0, stream>>>(X, Y, partial);
    }

    colw_kernel<<<(B_ * S_) / 256, 256, 0, stream>>>(partial, colw);
    out_kernel <<<(B_ * D_) / 256, 256, 0, stream>>>(X, colw, out + (size_t)B_ * S_);
}
